// ResNet_76166950028418
// MI455X (gfx1250) — matrix operand order repaired and checked by reading, not compiled, not run on hardware
//
#include <hip/hip_runtime.h>
#include <stdint.h>

// ---------------------------------------------------------------------------
// SNN block for MI455X (gfx1250, wave32).
// Only heavy stage: lin[t,b,o] = sum_i act[t,b,i] * fc_w[o,i]  (M=2048,K=7200,
// N=512, 15.1 GFLOP). Done with v_wmma_f32_16x16x32_bf16, f32 accumulate.
// Both operands are pre-swizzled into the contiguous-k (k = 16*half + j) WMMA fragment
// layouts so the GEMM kernel is LDS-free with fully coalesced 32B/lane loads.
// GEMM inner loop is a depth-2 software pipeline: fragment loads for kt+2 are
// in flight while the WMMAs of kt execute (two iterations of L2 latency
// coverage; no prefetch — WGP-scope asm prefetch was shown to break the
// scheduler, and SE/DEV prefetch stops at GL2 where data already resides).
// ---------------------------------------------------------------------------

typedef __attribute__((ext_vector_type(16))) __bf16 v16bf;
typedef __attribute__((ext_vector_type(8)))  float  v8f;

#define HW        120
#define CHANS     2
#define BATCH     256
#define TSTEPS    8
#define K_DIM     7200      // 2*60*60 pooled features
#define N_DIM     512
#define M_DIM     2048      // T*B
#define KT_CNT    225       // K / 32
#define NP_CNT    32        // N / 16
#define POOLW     60

// workspace byte offsets (all >= 16B aligned)
#define WS_CNT_OFF    0          // 2 x uint   (per-channel spike popcount)
#define WS_CONST_OFF  16         // 4 x float  (a0,a1,bb0,bb1)
#define WS_SPIKE_OFF  64         // 7,372,800 B  spike bitmasks [B][C][H][W]
#define WS_AFRAG_OFF  7372864    // 29,491,200 B A fragments (bf16)
#define WS_WFRAG_OFF  36864064   // 7,372,800 B  W fragments (bf16)
#define WS_LIN_OFF    44236864   // 4,194,304 B  lin f32 [M][N]

__device__ __forceinline__ unsigned short f2bf(float f) {
    unsigned u = __float_as_uint(f);
    u += 0x7FFFu + ((u >> 16) & 1u);      // round-to-nearest-even
    return (unsigned short)(u >> 16);
}

// --- zero the popcount counters -------------------------------------------
__global__ void k_init(unsigned* cnt) {
    if (threadIdx.x < 2) cnt[threadIdx.x] = 0u;
}

// --- conv3x3 (SAME) + bias + 8-step LIF (constant drive) + channel popcount
__global__ void k_conv_lif(const float* __restrict__ x,
                           const float* __restrict__ cw,
                           const float* __restrict__ cb,
                           unsigned char* __restrict__ spikes,
                           unsigned* __restrict__ cnt) {
    __shared__ unsigned sh[2];
    if (threadIdx.x < 2) sh[threadIdx.x] = 0u;
    __syncthreads();

    const int idx = blockIdx.x * 256 + threadIdx.x;   // < 7,372,800
    const int w = idx % HW;
    const int h = (idx / HW) % HW;
    const int c = (idx / (HW * HW)) % CHANS;
    const int b = idx / (HW * HW * CHANS);

    const float* xb = x + (size_t)b * HW * HW;        // single input channel
    float y = cb[c];
#pragma unroll
    for (int ky = 0; ky < 3; ++ky) {
#pragma unroll
        for (int kx = 0; kx < 3; ++kx) {
            const int hh = h + ky - 1, ww = w + kx - 1;
            if (hh >= 0 && hh < HW && ww >= 0 && ww < HW)
                y += cw[c * 9 + ky * 3 + kx] * xb[hh * HW + ww];
        }
    }
    // LIF: v = (v+y)/2 ; spike if v>=1 ; reset
    float v = 0.f; unsigned m = 0u;
#pragma unroll
    for (int t = 0; t < TSTEPS; ++t) {
        v = 0.5f * (v + y);
        if (v >= 1.0f) { m |= 1u << t; v = 0.f; }
    }
    spikes[idx] = (unsigned char)m;

    atomicAdd(&sh[c], (unsigned)__popc(m));           // ds_add
    __syncthreads();
    if (threadIdx.x < 2 && sh[threadIdx.x]) atomicAdd(&cnt[threadIdx.x], sh[threadIdx.x]);
}

// --- fold BN into per-channel affine: s_hat = a_c * s + bb_c ---------------
__global__ void k_bn_consts(const unsigned* __restrict__ cnt,
                            const float* __restrict__ gamma,
                            const float* __restrict__ beta,
                            float* __restrict__ cst) {
    const int c = threadIdx.x;
    if (c < CHANS) {
        const float N    = (float)TSTEPS * BATCH * HW * HW;   // 29,491,200
        const float mean = (float)cnt[c] / N;
        const float var  = mean - mean * mean;                // spikes in {0,1}
        const float a    = gamma[c] * rsqrtf(var + 1e-5f);
        cst[c]         = a;
        cst[2 + c]     = beta[c] - mean * a;
    }
}

// --- pack fc_w (f32 [512][7200]) into B-fragment layout (bf16) -------------
// chunk index: ((kt*32 + np)*32 + lane)*16 + j ; k_local = j + 16*(lane>>4)
__global__ void k_pack_w(const float* __restrict__ fcw, unsigned* __restrict__ Wf) {
    const int idx = blockIdx.x * 256 + threadIdx.x;   // one uint = 2 bf16; < 1,843,200
    const int jp   = idx & 7;
    const int lane = (idx >> 3) & 31;
    const int np   = (idx >> 8) & 31;
    const int kt   = idx >> 13;
    const int n    = np * 16 + (lane & 15);
    const int kb   = kt * 32 + 16 * (lane >> 4) + 2 * jp;
    const unsigned short lo = f2bf(fcw[(size_t)n * K_DIM + kb]);
    const unsigned short hi = f2bf(fcw[(size_t)n * K_DIM + kb + 1]);
    Wf[idx] = (unsigned)lo | ((unsigned)hi << 16);
}

// --- pack activations: spikes -> pooled -> BN affine -> A-fragment bf16 ----
// chunk index: ((mt*KT + kt)*32 + lane)*16 + j
// k_local = j + 16*(lane>>4) ;  row m = mt*16 + (lane&15)
__global__ void k_pack_a(const unsigned char* __restrict__ spikes,
                         const float* __restrict__ cst,
                         unsigned* __restrict__ Af) {
    const int idx = blockIdx.x * 256 + threadIdx.x;   // one uint = 2 bf16; < 7,372,800
    const int jp   = idx & 7;
    const int lane = (idx >> 3) & 31;
    const int kt   = (idx >> 8) % KT_CNT;
    const int mt   = (idx >> 8) / KT_CNT;
    const int half = lane >> 4;
    const int m    = mt * 16 + (lane & 15);
    const int t    = m >> 8;                           // row-major: tb = t*256+b
    const int b    = m & 255;

    unsigned out = 0u;
#pragma unroll
    for (int s = 0; s < 2; ++s) {
        const int j  = 2 * jp + s;
        const int kl = j + 16 * half;
        const int k  = kt * 32 + kl;
        const int c  = k / 3600;
        const int r  = k - c * 3600;
        const int pi = r / POOLW, pj = r - pi * POOLW;
        const size_t base = (((size_t)(b * CHANS + c) * HW) + 2 * pi) * HW + 2 * pj;
        const int q = ((spikes[base]          >> t) & 1) + ((spikes[base + 1]      >> t) & 1)
                    + ((spikes[base + HW]     >> t) & 1) + ((spikes[base + HW + 1] >> t) & 1);
        const float val = cst[c] * (0.25f * (float)q) + cst[2 + c];
        out |= (unsigned)f2bf(val) << (16 * s);
    }
    Af[idx] = out;
}

// --- WMMA GEMM: lin[M][N] = Af (M x K, bf16) * Wf (K x N, bf16), f32 accum -
// 256 threads = 8 waves; wave -> one 16-col N-panel, 4 M-tiles (64x16 out).
// Depth-2 software pipeline: loads for kt+2 issued before the WMMAs of kt.
__global__ void k_fc_gemm(const __bf16* __restrict__ Afb,
                          const __bf16* __restrict__ Wfb,
                          float* __restrict__ lin) {
    const int lane = threadIdx.x & 31;
    const int wave = threadIdx.x >> 5;            // 0..7
    const int np   = blockIdx.y * 8 + wave;       // 0..31
    const int mt0  = blockIdx.x * 4;              // 0..124

    const int MTSTRIDE = KT_CNT * 32;             // v16bf chunks per M-tile row
    const v16bf* ap = (const v16bf*)Afb + (size_t)mt0 * MTSTRIDE + lane;
    const v16bf* bp = (const v16bf*)Wfb + (size_t)np * 32 + lane;

    v8f c0 = {}, c1 = {}, c2 = {}, c3 = {};

    // ---- prologue: preload fragment sets for kt = 0 ("cur") and kt = 1 ("nxt")
    v16bf bf_c = bp[0];
    v16bf a0_c = ap[0];
    v16bf a1_c = ap[MTSTRIDE];
    v16bf a2_c = ap[2 * MTSTRIDE];
    v16bf a3_c = ap[3 * MTSTRIDE];
    ap += 32; bp += NP_CNT * 32;
    v16bf bf_n = bp[0];
    v16bf a0_n = ap[0];
    v16bf a1_n = ap[MTSTRIDE];
    v16bf a2_n = ap[2 * MTSTRIDE];
    v16bf a3_n = ap[3 * MTSTRIDE];

#pragma unroll 2
    for (int kt = 0; kt < KT_CNT - 2; ++kt) {
        ap += 32; bp += NP_CNT * 32;              // now points at kt+2
        // issue loads for kt+2 (stay in flight across this AND the next step)
        const v16bf bf_l = bp[0];
        const v16bf a0_l = ap[0];
        const v16bf a1_l = ap[MTSTRIDE];
        const v16bf a2_l = ap[2 * MTSTRIDE];
        const v16bf a3_l = ap[3 * MTSTRIDE];
        // compute kt (loaded two steps ago -> no wait on recent loads)
        c0 = __builtin_amdgcn_wmma_f32_16x16x32_bf16(false, a0_c, false, bf_c, (short)0, c0, false, false);
        c1 = __builtin_amdgcn_wmma_f32_16x16x32_bf16(false, a1_c, false, bf_c, (short)0, c1, false, false);
        c2 = __builtin_amdgcn_wmma_f32_16x16x32_bf16(false, a2_c, false, bf_c, (short)0, c2, false, false);
        c3 = __builtin_amdgcn_wmma_f32_16x16x32_bf16(false, a3_c, false, bf_c, (short)0, c3, false, false);
        // rotate pipeline registers (renamed away by the x2 unroll)
        bf_c = bf_n; a0_c = a0_n; a1_c = a1_n; a2_c = a2_n; a3_c = a3_n;
        bf_n = bf_l; a0_n = a0_l; a1_n = a1_l; a2_n = a2_l; a3_n = a3_l;
    }
    // ---- epilogue: kt = 223 and kt = 224 ----
    c0 = __builtin_amdgcn_wmma_f32_16x16x32_bf16(false, a0_c, false, bf_c, (short)0, c0, false, false);
    c1 = __builtin_amdgcn_wmma_f32_16x16x32_bf16(false, a1_c, false, bf_c, (short)0, c1, false, false);
    c2 = __builtin_amdgcn_wmma_f32_16x16x32_bf16(false, a2_c, false, bf_c, (short)0, c2, false, false);
    c3 = __builtin_amdgcn_wmma_f32_16x16x32_bf16(false, a3_c, false, bf_c, (short)0, c3, false, false);
    c0 = __builtin_amdgcn_wmma_f32_16x16x32_bf16(false, a0_n, false, bf_n, (short)0, c0, false, false);
    c1 = __builtin_amdgcn_wmma_f32_16x16x32_bf16(false, a1_n, false, bf_n, (short)0, c1, false, false);
    c2 = __builtin_amdgcn_wmma_f32_16x16x32_bf16(false, a2_n, false, bf_n, (short)0, c2, false, false);
    c3 = __builtin_amdgcn_wmma_f32_16x16x32_bf16(false, a3_n, false, bf_n, (short)0, c3, false, false);

    // C/D layout: lanes 0-15 rows v, lanes 16-31 rows v+8; col = np*16 + lane%16
    const int rbase = (lane >> 4) * 8;
    const int n     = np * 16 + (lane & 15);
#pragma unroll
    for (int v = 0; v < 8; ++v) {
        lin[(size_t)((mt0 + 0) * 16 + rbase + v) * N_DIM + n] = c0[v];
        lin[(size_t)((mt0 + 1) * 16 + rbase + v) * N_DIM + n] = c1[v];
        lin[(size_t)((mt0 + 2) * 16 + rbase + v) * N_DIM + n] = c2[v];
        lin[(size_t)((mt0 + 3) * 16 + rbase + v) * N_DIM + n] = c3[v];
    }
}

// --- second LIF over time-varying drive + firing-rate mean -----------------
__global__ void k_lif2(const float* __restrict__ lin,
                       const float* __restrict__ fcb,
                       float* __restrict__ out) {
    const int idx = blockIdx.x * 256 + threadIdx.x;   // < 131072
    const int o = idx & (N_DIM - 1);
    const int b = idx >> 9;
    const float bias = fcb[o];
    float v = 0.f; int cnt = 0;
#pragma unroll
    for (int t = 0; t < TSTEPS; ++t) {
        const float xin = lin[(size_t)(t * BATCH + b) * N_DIM + o] + bias;
        v = 0.5f * (v + xin);
        if (v >= 1.0f) { ++cnt; v = 0.f; }
    }
    out[idx] = (float)cnt * 0.125f;                   // fr[b][o]
}

extern "C" void kernel_launch(void* const* d_in, const int* in_sizes, int n_in,
                              void* d_out, int out_size, void* d_ws, size_t ws_size,
                              hipStream_t stream) {
    const float* x     = (const float*)d_in[0];
    const float* cw    = (const float*)d_in[1];
    const float* cb    = (const float*)d_in[2];
    const float* gamma = (const float*)d_in[3];
    const float* beta  = (const float*)d_in[4];
    const float* fcw   = (const float*)d_in[5];
    const float* fcb   = (const float*)d_in[6];
    float* out = (float*)d_out;

    char* ws = (char*)d_ws;
    unsigned*       cnt    = (unsigned*)(ws + WS_CNT_OFF);
    float*          cst    = (float*)(ws + WS_CONST_OFF);
    unsigned char*  spikes = (unsigned char*)(ws + WS_SPIKE_OFF);
    unsigned*       AfU    = (unsigned*)(ws + WS_AFRAG_OFF);
    unsigned*       WfU    = (unsigned*)(ws + WS_WFRAG_OFF);
    __bf16*         Afb    = (__bf16*)(ws + WS_AFRAG_OFF);
    __bf16*         Wfb    = (__bf16*)(ws + WS_WFRAG_OFF);
    float*          lin    = (float*)(ws + WS_LIN_OFF);

    k_init<<<1, 64, 0, stream>>>(cnt);
    k_conv_lif<<<(BATCH * CHANS * HW * HW) / 256, 256, 0, stream>>>(x, cw, cb, spikes, cnt);
    k_bn_consts<<<1, 32, 0, stream>>>(cnt, gamma, beta, cst);
    k_pack_w<<<(K_DIM * N_DIM / 2) / 256, 256, 0, stream>>>(fcw, WfU);
    k_pack_a<<<(M_DIM * K_DIM / 2) / 256, 256, 0, stream>>>(spikes, cst, AfU);
    k_fc_gemm<<<dim3(32, 4), 256, 0, stream>>>(Afb, Wfb, lin);
    k_lif2<<<(BATCH * N_DIM) / 256, 256, 0, stream>>>(lin, fcb, out);
}
